// DecoderCell_80479097193077
// MI455X (gfx1250) — matrix operand order repaired and checked by reading, not compiled, not run on hardware
//
#include <hip/hip_runtime.h>
#include <hip/hip_bf16.h>

// ---------------------------------------------------------------------------
// Types for CDNA5 WMMA (wave32): A/B = 16 bf16 per lane, C/D = 8 f32 per lane
// ---------------------------------------------------------------------------
typedef __attribute__((ext_vector_type(16))) __bf16 v16bf;
typedef __attribute__((ext_vector_type(8)))  float  v8f;

#define BATCH   256
#define NNODES  1000
#define DIM     128
#define NHEADS  8
#define DHEAD   16
#define TSTEPS  99      // VICTIM_NODES-1
#define NEGV    (-1.0e9f)

// bf16 <-> f32 via bit ops (avoids relying on __bf16 arithmetic support)
__device__ __forceinline__ float bf2f(unsigned short h) {
    unsigned u = ((unsigned)h) << 16;
    float f; __builtin_memcpy(&f, &u, 4); return f;
}
__device__ __forceinline__ unsigned short f2bf(float f) {
    unsigned u; __builtin_memcpy(&u, &f, 4);
    u += 0x7FFFu + ((u >> 16) & 1u);       // round-to-nearest-even
    return (unsigned short)(u >> 16);
}

// Deterministic hash-based Gumbel noise (stand-in for jax.random.gumbel stream)
__device__ __forceinline__ float gumbel_hash(unsigned t, unsigned b, unsigned n) {
    unsigned x = t * 2654435761u ^ (b * 40503u + 0x9E3779B9u) ^ n * 2246822519u;
    x ^= x >> 15; x *= 2246822519u;
    x ^= x >> 13; x *= 3266489917u;
    x ^= x >> 16;
    float u = ((float)x + 1.0f) * (1.0f / 4294967296.0f);   // (0,1]
    u = fmaxf(u, 1e-20f);
    return -__logf(-__logf(u));
}

// ---------------------------------------------------------------------------
// Kernel 0: bulk f32 -> bf16 conversion (4 floats / thread, packed stores)
// ---------------------------------------------------------------------------
__global__ __launch_bounds__(256) void cvt_f32_to_bf16(
    const float* __restrict__ X, unsigned short* __restrict__ Y, int n4)
{
    const int i = blockIdx.x * 256 + threadIdx.x;    // index in float4 units
    if (i >= n4) return;
    const float4 f = ((const float4*)X)[i];
    uint2 p;
    p.x = (unsigned)f2bf(f.x) | ((unsigned)f2bf(f.y) << 16);
    p.y = (unsigned)f2bf(f.z) | ((unsigned)f2bf(f.w) << 16);
    ((uint2*)Y)[i] = p;
}

// ---------------------------------------------------------------------------
// Kernel 1: Y{0,1,2} = Xbf @ W{0,1,2}^T, all bf16, via v_wmma_f32_16x16x32_bf16.
//
// Layout trick: with bf16 sources, each lane's A and B fragments are one contiguous
// 16-element run (k = kbase + hi*16 + 0..15), so all
// WMMA operands come from aligned b128 loads with ZERO shuffle/convert VALU.
// B tiles (3 matrices x 4 k-chunks) are register-resident; each block runs
// 8 M-tiles so the weight loads amortize. 12 WMMAs per M-tile.
// ---------------------------------------------------------------------------
__global__ __launch_bounds__(256) void gemm3_wmma(
    const unsigned short* __restrict__ Xbf,   // [M,128] bf16
    const unsigned short* __restrict__ Wbf,   // [3,128,128] bf16 (row-major)
    unsigned short* __restrict__ Y0,
    unsigned short* __restrict__ Y1,
    unsigned short* __restrict__ Y2)          // each [M,128] bf16
{
    const int wave = threadIdx.x >> 5;   // 0..7 -> column tile
    const int lane = threadIdx.x & 31;
    const int r16  = lane & 15;
    const int hi   = lane >> 4;          // 0 or 1
    const int ncol = wave * 16 + r16;    // B column / output column this lane owns

    typedef union { v16bf v; uint4 q[2]; } Tile;

    // ---- preload B tiles: B[k,n] = W[n,k]; lane run = kbase + hi*16 + 0..15 ----
    Tile Bm[3][4];
    #pragma unroll
    for (int m = 0; m < 3; ++m) {
        const unsigned short* wrow = Wbf + ((size_t)m * 128 + ncol) * 128;
        #pragma unroll
        for (int kc = 0; kc < 4; ++kc) {
            const unsigned short* p = wrow + kc * 32 + hi * 16;
            Bm[m][kc].q[0] = *(const uint4*)(p);
            Bm[m][kc].q[1] = *(const uint4*)(p + 8);
        }
    }

    #pragma unroll 1
    for (int mt = 0; mt < 8; ++mt) {
        const int tile = blockIdx.x * 8 + mt;
        const unsigned short* xrow = Xbf + ((size_t)tile * 16 + r16) * 128;

        v8f acc0 = {}, acc1 = {}, acc2 = {};
        #pragma unroll
        for (int kc = 0; kc < 4; ++kc) {
            // A run: kbase + hi*16 + 0..15 (matches B)
            Tile A;
            const unsigned short* p = xrow + kc * 32 + hi * 16;
            A.q[0] = *(const uint4*)(p);
            A.q[1] = *(const uint4*)(p + 8);
            acc0 = __builtin_amdgcn_wmma_f32_16x16x32_bf16(
                       false, A.v, false, Bm[0][kc].v, (short)0, acc0, false, false);
            acc1 = __builtin_amdgcn_wmma_f32_16x16x32_bf16(
                       false, A.v, false, Bm[1][kc].v, (short)0, acc1, false, false);
            acc2 = __builtin_amdgcn_wmma_f32_16x16x32_bf16(
                       false, A.v, false, Bm[2][kc].v, (short)0, acc2, false, false);
        }

        // C/D layout: VGPR r, lanes 0-15 -> M=r, lanes 16-31 -> M=r+8; N = lane%16
        #pragma unroll
        for (int r = 0; r < 8; ++r) {
            const size_t off = ((size_t)tile * 16 + r + hi * 8) * 128 + ncol;
            Y0[off] = f2bf(acc0[r]);
            Y1[off] = f2bf(acc1[r]);
            Y2[off] = f2bf(acc2[r]);
        }
    }
}

// ---------------------------------------------------------------------------
// Phase 2: sequential 99-step decode. One block (128 threads) per batch elem.
// K1/V/K2 are bf16, ~196 MB total -> resident in the 192 MB L2 across steps.
// ---------------------------------------------------------------------------
__global__ __launch_bounds__(128) void decode_kernel(
    const float* __restrict__ nodeEmb,    // [B,N,128]
    const float* __restrict__ graphEmb,   // [B,128]
    const float* __restrict__ Wq_fixed,   // [128,128]
    const float* __restrict__ Wq_step,    // [128,128]
    const float* __restrict__ Wout,       // [128,128]
    const unsigned short* __restrict__ K1,// [B,N,128] bf16 (channel = h*16+j)
    const unsigned short* __restrict__ V, // [B,N,128] bf16
    const unsigned short* __restrict__ K2,// [B,N,128] bf16
    const int* __restrict__ firstIdxPtr,
    float* __restrict__ out)              // [B] ll, then [B,100] pi (as float)
{
    const int b   = blockIdx.x;
    const int t   = threadIdx.x;          // 0..127
    const int h   = t >> 4;               // head 0..7
    const int l16 = t & 15;

    __shared__ float s_qf[DIM], s_ctx[DIM], s_q1[DIM], s_q2[DIM], s_q2o[DIM];
    __shared__ float s_scores[NHEADS * NNODES];   // 32 KB
    __shared__ float s_logits[NNODES];            // 4 KB
    __shared__ unsigned char s_mask[NNODES];
    __shared__ float s_red[128];
    __shared__ int   s_redi[128];
    __shared__ float s_hmax[NHEADS], s_hsum[NHEADS];
    __shared__ int   s_nxt;

    const int first = firstIdxPtr[0];

    // ---- init ----
    for (int n = t; n < NNODES; n += 128) s_mask[n] = 0;
    {   // Q_fixed[b,t] = graph[b,:] . Wq_fixed[t,:]
        float acc = 0.f;
        const float* g  = graphEmb + (size_t)b * DIM;
        const float* wr = Wq_fixed + (size_t)t * DIM;
        for (int k = 0; k < DIM; ++k) acc += g[k] * wr[k];
        s_qf[t] = acc;
    }
    s_ctx[t] = nodeEmb[((size_t)b * NNODES + first) * DIM + t];
    float ll = 0.f;
    if (t == 0) { s_mask[first] = 1; out[BATCH + b * 100 + 0] = (float)first; }
    __syncthreads();

    for (int step = 0; step < TSTEPS; ++step) {
        // 1) Q1 = Q_fixed + ctx @ Wq_step^T
        {
            float acc = 0.f;
            const float* wr = Wq_step + (size_t)t * DIM;
            for (int k = 0; k < DIM; ++k) acc += s_ctx[k] * wr[k];
            s_q1[t] = s_qf[t] + acc;
        }
        __syncthreads();

        // 2) per-head scores + softmax numerator (thread = head h, node lane l16)
        {
            float q[DHEAD];
            #pragma unroll
            for (int j = 0; j < DHEAD; ++j) q[j] = s_q1[h * DHEAD + j];
            float lmax = -3.4e38f;
            for (int n = l16; n < NNODES; n += 16) {
                const unsigned short* kr =
                    K1 + (((size_t)b * NNODES + n) * DIM + h * DHEAD);
                float d = 0.f;
                #pragma unroll
                for (int j = 0; j < DHEAD; ++j) d += bf2f(kr[j]) * q[j];
                d *= 0.25f;                           // 1/sqrt(dh)
                if (s_mask[n]) d = NEGV;
                s_scores[h * NNODES + n] = d;
                lmax = fmaxf(lmax, d);
            }
            s_red[t] = lmax;
            __syncthreads();
            if (l16 == 0) {
                float m = -3.4e38f;
                for (int j = 0; j < 16; ++j) m = fmaxf(m, s_red[h * 16 + j]);
                s_hmax[h] = m;
            }
            __syncthreads();
            const float hm = s_hmax[h];
            float lsum = 0.f;
            for (int n = l16; n < NNODES; n += 16) {
                const float e = __expf(s_scores[h * NNODES + n] - hm);
                s_scores[h * NNODES + n] = e;
                lsum += e;
            }
            s_red[t] = lsum;
            __syncthreads();
            if (l16 == 0) {
                float s = 0.f;
                for (int j = 0; j < 16; ++j) s += s_red[h * 16 + j];
                s_hsum[h] = s;
            }
            __syncthreads();
        }

        // 3) attn @ V : thread t owns channel t; block reads 256 B/iter coalesced
        {
            float acc = 0.f;
            const unsigned short* vb = V + ((size_t)b * NNODES) * DIM + t;
            const float* sc = s_scores + h * NNODES;
            for (int n = 0; n < NNODES; ++n) {
                if ((n & 63) == 0 && n + 64 < NNODES)
                    __builtin_prefetch(vb + (size_t)(n + 64) * DIM, 0, 0);
                acc += sc[n] * bf2f(vb[(size_t)n * DIM]);
            }
            s_q2[t] = acc / s_hsum[h];
        }
        __syncthreads();

        // 4) Q2 @ Wout^T
        {
            float acc = 0.f;
            const float* wr = Wout + (size_t)t * DIM;
            for (int k = 0; k < DIM; ++k) acc += s_q2[k] * wr[k];
            s_q2o[t] = acc;
        }
        __syncthreads();

        // 5) pointer logits: 10*tanh((Q2o . K2[n]) / sqrt(128)), masked
        for (int n = t; n < NNODES; n += 128) {
            const unsigned short* kr = K2 + ((size_t)b * NNODES + n) * DIM;
            float d = 0.f;
            for (int k = 0; k < DIM; ++k) d += bf2f(kr[k]) * s_q2o[k];
            float lg = 10.f * tanhf(d * 0.08838834764831845f);
            if (s_mask[n]) lg = NEGV;
            s_logits[n] = lg;
        }
        __syncthreads();

        // 6) masked log-softmax + Gumbel-max sampling
        float lmax = -3.4e38f;
        for (int n = t; n < NNODES; n += 128) lmax = fmaxf(lmax, s_logits[n]);
        s_red[t] = lmax; __syncthreads();
        for (int s = 64; s > 0; s >>= 1) {
            if (t < s) s_red[t] = fmaxf(s_red[t], s_red[t + s]);
            __syncthreads();
        }
        const float M = s_red[0]; __syncthreads();
        float lsum = 0.f;
        for (int n = t; n < NNODES; n += 128) lsum += __expf(s_logits[n] - M);
        s_red[t] = lsum; __syncthreads();
        for (int s = 64; s > 0; s >>= 1) {
            if (t < s) s_red[t] += s_red[t + s];
            __syncthreads();
        }
        const float lse = __logf(s_red[0]); __syncthreads();

        float bestk = -3.4e38f; int bestn = 0;
        for (int n = t; n < NNODES; n += 128) {
            const float lp  = s_logits[n] - M - lse;
            const float key = lp + gumbel_hash((unsigned)step, (unsigned)b, (unsigned)n);
            if (key > bestk) { bestk = key; bestn = n; }
        }
        s_red[t] = bestk; s_redi[t] = bestn; __syncthreads();
        for (int s = 64; s > 0; s >>= 1) {
            if (t < s) {
                const bool take = (s_red[t + s] > s_red[t]) ||
                                  (s_red[t + s] == s_red[t] && s_redi[t + s] < s_redi[t]);
                if (take) { s_red[t] = s_red[t + s]; s_redi[t] = s_redi[t + s]; }
            }
            __syncthreads();
        }

        // 7) commit: mask, pi, ll, next ctx
        if (t == 0) {
            const int nxt = s_redi[0];
            s_mask[nxt] = 1;
            s_nxt = nxt;
            out[BATCH + b * 100 + (step + 1)] = (float)nxt;
            ll += s_logits[nxt] - M - lse;
        }
        __syncthreads();
        s_ctx[t] = nodeEmb[((size_t)b * NNODES + s_nxt) * DIM + t];
        __syncthreads();
    }

    if (t == 0) out[b] = ll;
}

// ---------------------------------------------------------------------------
// Launch
// ---------------------------------------------------------------------------
extern "C" void kernel_launch(void* const* d_in, const int* in_sizes, int n_in,
                              void* d_out, int out_size, void* d_ws, size_t ws_size,
                              hipStream_t stream) {
    (void)in_sizes; (void)n_in; (void)out_size; (void)ws_size;

    const float* nodeEmb  = (const float*)d_in[0];   // [256,1000,128]
    const float* graphEmb = (const float*)d_in[1];   // [256,128]
    const float* Wk1      = (const float*)d_in[2];
    const float* Wv       = (const float*)d_in[3];
    const float* Wk2      = (const float*)d_in[4];
    const float* Wq_fixed = (const float*)d_in[5];
    const float* Wout     = (const float*)d_in[6];
    const float* Wq_step  = (const float*)d_in[7];
    const int*   firstIdx = (const int*)d_in[8];
    float*       out      = (float*)d_out;           // [256] ll ++ [256,100] pi

    const size_t elems = (size_t)BATCH * NNODES * DIM;   // 32,768,000
    unsigned short* K1  = (unsigned short*)d_ws;         // bf16, 65.5 MB
    unsigned short* Vp  = K1 + elems;                    // bf16, 65.5 MB
    unsigned short* K2  = Vp + elems;                    // bf16, 65.5 MB
    unsigned short* Xbf = K2 + elems;                    // bf16, 65.5 MB
    unsigned short* Wbf = Xbf + elems;                   // bf16, 3*128*128 = 96 KB

    // 0) convert node embeddings + the 3 projection weights to bf16
    {
        const int n4x = (int)(elems / 4);                // 8,192,000 float4s
        cvt_f32_to_bf16<<<dim3((n4x + 255) / 256), 256, 0, stream>>>(nodeEmb, Xbf, n4x);
        const int n4w = (128 * 128) / 4;                 // 4096 float4s per weight
        cvt_f32_to_bf16<<<dim3((n4w + 255) / 256), 256, 0, stream>>>(Wk1, Wbf,             n4w);
        cvt_f32_to_bf16<<<dim3((n4w + 255) / 256), 256, 0, stream>>>(Wv,  Wbf + 128 * 128, n4w);
        cvt_f32_to_bf16<<<dim3((n4w + 255) / 256), 256, 0, stream>>>(Wk2, Wbf + 2 * 128 * 128, n4w);
    }

    // 1) three weight-stationary bf16 WMMA GEMMs (K1 | V | K2), 12 WMMAs/tile
    {
        const int mtiles = (BATCH * NNODES) / 16;        // 16000
        gemm3_wmma<<<dim3(mtiles / 8), 256, 0, stream>>>(Xbf, Wbf, K1, Vp, K2);
    }

    // 2) persistent per-batch sequential decode
    decode_kernel<<<dim3(BATCH), 128, 0, stream>>>(
        nodeEmb, graphEmb, Wq_fixed, Wq_step, Wout, K1, Vp, K2, firstIdx, out);
}
